// HydroGATLayer_24704651887253
// MI455X (gfx1250) — matrix operand order repaired and checked by reading, not compiled, not run on hardware
//
#include <hip/hip_runtime.h>
#include <hip/hip_bf16.h>
#include <math.h>

typedef _Float16 v16h __attribute__((ext_vector_type(16)));
typedef float    v8f  __attribute__((ext_vector_type(8)));

#define Bn   8
#define Nn   1024
#define FIN  64
#define FOUT 128
#define Hn   4
#define Dn   32
#define NEG_SLOPE 0.2f

__device__ __forceinline__ float leaky(float v) { return v >= 0.f ? v : NEG_SLOPE * v; }

// ---------------------------------------------------------------------------
// Kernel 1: h = x @ W   (per-wave 16x16 tile, two K=32 WMMA steps)
// Stores h as f32 [b,n,128] and transposed f16 [b,feat,n] for aggregation.
// ---------------------------------------------------------------------------
__global__ void k_gemm_h(const float* __restrict__ x, const float* __restrict__ W,
                         float* __restrict__ hfeat, _Float16* __restrict__ hT16)
{
    const int wid   = blockIdx.x;          // Bn * 64 * 8 waves
    const int ntile = wid & 7;             // feature tile (8 x 16 = 128)
    const int mtile = (wid >> 3) & 63;     // node tile (64 x 16 = 1024)
    const int b     = wid >> 9;
    const int lane  = threadIdx.x;         // 0..31
    const int lrow  = lane & 15;
    const int hi    = lane >> 4;

    v8f acc = {};
    const int   arow = mtile * 16 + lrow;
    const float* xrow = x + (size_t)(b * Nn + arow) * FIN;
    const int   ncol = ntile * 16 + lrow;

    #pragma unroll
    for (int chunk = 0; chunk < 2; ++chunk) {
        // A operand: 16x32 f16. lane<16: K=base..base+7 then base+16..base+23
        //            lane>=16: same with base+8.
        const int kbase = chunk * 32 + hi * 8;
        const float* p0 = xrow + kbase;
        const float* p1 = xrow + kbase + 16;
        v16h A;
        #pragma unroll
        for (int i2 = 0; i2 < 8; ++i2) {
            A[i2]     = (_Float16)p0[i2];
            A[8 + i2] = (_Float16)p1[i2];
        }
        // B operand: 32x16 f16. lane<16: N=lane, K=kb..kb+7 then kb+16..kb+23; lane>=16: kb+=8.
        const int kb = chunk * 32 + hi * 8;
        v16h Bv;
        #pragma unroll
        for (int i2 = 0; i2 < 16; ++i2)
            Bv[i2] = (_Float16)W[(kb + i2 + (i2 & 8)) * FOUT + ncol];

        acc = __builtin_amdgcn_wmma_f32_16x16x32_f16(false, A, false, Bv,
                                                     (short)0, acc, false, false);
    }

    // C/D layout: VGPR r -> M = r (lanes 0-15) or 8+r (lanes 16-31), N = lane&15
    #pragma unroll
    for (int r = 0; r < 8; ++r) {
        const int row = mtile * 16 + hi * 8 + r;
        const float vv = acc[r];
        hfeat[(size_t)(b * Nn + row) * FOUT + ncol] = vv;
        hT16[(size_t)(b * FOUT + ncol) * Nn + row]  = (_Float16)vv;   // contiguous in r
    }
}

// ---------------------------------------------------------------------------
// Kernel 2: e_src/e_dst per (b, head, node), stored transposed [b,h,n]
// ---------------------------------------------------------------------------
__global__ void k_edges(const float* __restrict__ hfeat, const float* __restrict__ a,
                        float* __restrict__ esrcT, float* __restrict__ edstT)
{
    const int t = blockIdx.x * blockDim.x + threadIdx.x;
    if (t >= Bn * Hn * Nn) return;
    const int n = t & (Nn - 1);
    const int h = (t >> 10) & (Hn - 1);
    const int b = t >> 12;
    const float* hv = hfeat + (size_t)(b * Nn + n) * FOUT + h * Dn;
    const float* as = a + h * 2 * Dn;
    float s0 = 0.f, s1 = 0.f;
    #pragma unroll 8
    for (int d = 0; d < Dn; ++d) {
        const float v = hv[d];
        s0 += v * as[d];
        s1 += v * as[Dn + d];
    }
    esrcT[t] = s0;   // index == (b*Hn+h)*Nn + n
    edstT[t] = s1;
}

// ---------------------------------------------------------------------------
// Kernel 3: per-row softmax stats (online max/sum), one wave per (b,row).
// 8 waves per 256-thread block -> Bn*Nn/8 = 1024 blocks, 8192 waves.
// ---------------------------------------------------------------------------
__global__ void k_stats(const int* __restrict__ adj,
                        const float* __restrict__ esrcT, const float* __restrict__ edstT,
                        float* __restrict__ mS, float* __restrict__ invS)
{
    const int row  = blockIdx.x * 8 + (threadIdx.x >> 5);  // 0 .. Bn*Nn-1
    const int lane = threadIdx.x & 31;
    const int b    = row >> 10;
    const int i    = row & (Nn - 1);
    const float NEG_INF = -__builtin_inff();

    float es[Hn], m[Hn], s[Hn];
    #pragma unroll
    for (int h = 0; h < Hn; ++h) {
        es[h] = esrcT[(b * Hn + h) * Nn + i];
        m[h]  = NEG_INF;
        s[h]  = 0.f;
    }

    const int* arow = adj + (size_t)i * Nn;
    for (int t = 0; t < Nn / 32; ++t) {
        const int j  = lane + t * 32;
        const int av = arow[j];
        if (t + 1 < Nn / 32)
            __builtin_prefetch(arow + j + 32, 0, 1);
        #pragma unroll
        for (int h = 0; h < Hn; ++h) {
            const float edv = edstT[(b * Hn + h) * Nn + j];
            const float v   = leaky(es[h] + edv);
            if (av != 0) {
                if (v > m[h]) { s[h] = s[h] * __expf(m[h] - v) + 1.f; m[h] = v; }
                else          { s[h] += __expf(v - m[h]); }
            }
        }
    }

    // butterfly combine of (m,s) across the wave (wave32)
    #pragma unroll
    for (int h = 0; h < Hn; ++h) {
        float mm = m[h], ss = s[h];
        #pragma unroll
        for (int off = 16; off >= 1; off >>= 1) {
            const float mo = __shfl_xor(mm, off, 32);
            const float so = __shfl_xor(ss, off, 32);
            const float mn = fmaxf(mm, mo);
            float sn = 0.f;
            if (mn > NEG_INF) sn = ss * __expf(mm - mn) + so * __expf(mo - mn);
            mm = mn; ss = sn;
        }
        if (lane == h) {   // one distinct lane per head does the write
            mS  [(b * Hn + h) * Nn + i] = (mm > NEG_INF) ? mm : 0.f;
            invS[(b * Hn + h) * Nn + i] = (ss > 0.f) ? 1.f / ss : 0.f;
        }
    }
}

// ---------------------------------------------------------------------------
// Kernel 4: emit attn (WMMA A layout), write avg_attn, WMMA-accumulate out.
// 512 blocks of 4 waves; wave w owns j-tiles  w, w+4, w+8, ...  (8 each).
// Per-wave out partials are summed deterministically through LDS.
// ---------------------------------------------------------------------------
__global__ void k_attn(const int* __restrict__ adj, const float* __restrict__ ew,
                       const float* __restrict__ esrcT, const float* __restrict__ edstT,
                       const float* __restrict__ mS, const float* __restrict__ invS,
                       const _Float16* __restrict__ hT16,
                       float* __restrict__ out, float* __restrict__ avg)
{
    const int b     = blockIdx.x >> 6;
    const int itile = blockIdx.x & 63;
    const int tid   = threadIdx.x;
    const int wave  = tid >> 5;            // 0..3
    const int lane  = tid & 31;
    const int lrow  = lane & 15;
    const int hi    = lane >> 4;

    __shared__ float part[4][32][64];      // 32 KB: per-wave out partials

    const int i = itile * 16 + lrow;       // my A-matrix row
    float mrow[Hn], invrow[Hn], esrow[Hn];
    #pragma unroll
    for (int h = 0; h < Hn; ++h) {
        mrow[h]   = mS   [(b * Hn + h) * Nn + i];
        invrow[h] = invS [(b * Hn + h) * Nn + i];
        esrow[h]  = esrcT[(b * Hn + h) * Nn + i];
    }

    v8f acc[Hn][2];
    #pragma unroll
    for (int h = 0; h < Hn; ++h) { v8f z = {}; acc[h][0] = z; acc[h][1] = z; }

    const int  koff  = hi * 16;
    const int* arow  = adj + (size_t)i * Nn;
    const float* ewr = ew  + (size_t)i * Nn;
    float* avrow     = avg + (size_t)(b * Nn + i) * Nn;

    for (int jt = wave; jt < Nn / 32; jt += 4) {
        const int jbase = jt * 32;
        const int j0 = jbase + koff;       // A elements 0..7  -> K = koff+k
        const int j1 = j0 + 8;             // A elements 8..15 -> K = koff+8+k

        int   avv[16];
        float eww[16], avga[16];
        #pragma unroll
        for (int k = 0; k < 8; ++k) {
            avv[k]  = arow[j0 + k];  avv[8 + k] = arow[j1 + k];
            eww[k]  = ewr[j0 + k];   eww[8 + k] = ewr[j1 + k];
            avga[k] = 0.f;           avga[8 + k] = 0.f;
        }

        v16h A[Hn];
        #pragma unroll
        for (int h = 0; h < Hn; ++h) {
            const float* edp = edstT + (b * Hn + h) * Nn;
            float edv[16];
            #pragma unroll
            for (int k = 0; k < 8; ++k) { edv[k] = edp[j0 + k]; edv[8 + k] = edp[j1 + k]; }
            #pragma unroll
            for (int k = 0; k < 16; ++k) {
                float p = 0.f;
                if (avv[k])
                    p = __expf(leaky(esrow[h] + edv[k]) - mrow[h]) * invrow[h];
                const float w = p * eww[k];
                A[h][k] = (_Float16)w;
                avga[k] += w;
            }
        }

        // avg_attn: contiguous 8-float runs, each (i,j) written exactly once
        #pragma unroll
        for (int k = 0; k < 8; ++k) {
            avrow[j0 + k] = avga[k]     * 0.25f;
            avrow[j1 + k] = avga[8 + k] * 0.25f;
        }

        // B operand: hT16[b, feat, j] -> contiguous 16 halfs along K
        #pragma unroll
        for (int h = 0; h < Hn; ++h) {
            #pragma unroll
            for (int dt = 0; dt < 2; ++dt) {
                const _Float16* bp =
                    hT16 + (size_t)(b * FOUT + h * Dn + dt * 16 + lrow) * Nn + jbase + hi * 16;
                const v16h Bv = *(const v16h*)bp;
                acc[h][dt] = __builtin_amdgcn_wmma_f32_16x16x32_f16(
                    false, A[h], false, Bv, (short)0, acc[h][dt], false, false);
            }
        }
    }

    // dump per-wave partials, then reduce across the 4 waves deterministically
    #pragma unroll
    for (int h = 0; h < Hn; ++h)
        #pragma unroll
        for (int dt = 0; dt < 2; ++dt)
            #pragma unroll
            for (int r = 0; r < 8; ++r)
                part[wave][lane][(h * 2 + dt) * 8 + r] = acc[h][dt][r];
    __syncthreads();

    // out tile = 16 rows x 128 cols; 128 threads x 16 elements each
    for (int e = tid; e < 16 * FOUT; e += 128) {
        const int row  = e >> 7;
        const int col  = e & 127;
        const int l2   = ((row >> 3) << 4) | (col & 15);          // source lane
        const int idx  = (((col >> 5) * 2 + ((col >> 4) & 1)) << 3) | (row & 7);
        const float sv = part[0][l2][idx] + part[1][l2][idx]
                       + part[2][l2][idx] + part[3][l2][idx];
        out[(size_t)(b * Nn + itile * 16 + row) * FOUT + col] = sv;
    }
}

// ---------------------------------------------------------------------------
extern "C" void kernel_launch(void* const* d_in, const int* in_sizes, int n_in,
                              void* d_out, int out_size, void* d_ws, size_t ws_size,
                              hipStream_t stream)
{
    const float* x   = (const float*)d_in[0];   // [B,N,64]
    const int*   adj = (const int*)  d_in[1];   // [N,N]
    const float* ewp = (const float*)d_in[2];   // [N,N]
    const float* W   = (const float*)d_in[3];   // [64,128]
    const float* a   = (const float*)d_in[4];   // [H,2D]

    float* out = (float*)d_out;                       // [B,N,128]
    float* avg = out + (size_t)Bn * Nn * FOUT;        // [B,N,N]

    // ws layout: hfeat f32 4MB | hT16 f16 2MB | esrcT 128K | edstT 128K | mS 128K | invS 128K
    char* ws = (char*)d_ws;
    float*    hfeat = (float*)ws;
    _Float16* hT16  = (_Float16*)(ws + (size_t)Bn * Nn * FOUT * 4);
    float*    esrcT = (float*)(ws + (size_t)Bn * Nn * FOUT * 6);
    float*    edstT = esrcT + Bn * Hn * Nn;
    float*    mS    = edstT + Bn * Hn * Nn;
    float*    invS  = mS    + Bn * Hn * Nn;

    k_gemm_h<<<Bn * (Nn / 16) * (FOUT / 16), 32, 0, stream>>>(x, W, hfeat, hT16);
    k_edges <<<(Bn * Hn * Nn) / 256, 256, 0, stream>>>(hfeat, a, esrcT, edstT);
    k_stats <<<(Bn * Nn) / 8, 256, 0, stream>>>(adj, esrcT, edstT, mS, invS);
    k_attn  <<<Bn * (Nn / 16), 128, 0, stream>>>(adj, ewp, esrcT, edstT, mS, invS,
                                                 hT16, out, avg);
}
